// PPFNetLocalGlobal_317827580548
// MI455X (gfx1250) — matrix operand order repaired and checked by reading, not compiled, not run on hardware
//
#include <hip/hip_runtime.h>
#include <hip/hip_bf16.h>
#include <stdint.h>

// ---------------- problem constants (from reference) ----------------
#define BB   2
#define NN   16384
#define SS   512
#define KK   64
#define EMB  96
#define C2   192          // EMB*2
#define R2   0.09f        // RADIUS^2
#define EPS12 1e-12f

typedef __attribute__((ext_vector_type(16))) __bf16 v16bf;
typedef __attribute__((ext_vector_type(8)))  __bf16 v8bf;   // one 128-bit load
typedef __attribute__((ext_vector_type(8)))  float  v8f;

struct F3 { float x, y, z; };
__device__ __forceinline__ F3 ld3(const float* p) { return {p[0], p[1], p[2]}; }
__device__ __forceinline__ F3 f3sub(F3 a, F3 b) { return {a.x-b.x, a.y-b.y, a.z-b.z}; }
__device__ __forceinline__ float f3dot(F3 a, F3 b) { return a.x*b.x + a.y*b.y + a.z*b.z; }
__device__ __forceinline__ F3 f3cross(F3 a, F3 b) {
  return {a.y*b.z - a.z*b.y, a.z*b.x - a.x*b.z, a.x*b.y - a.y*b.x};
}
__device__ __forceinline__ float safenorm3(F3 a) { return sqrtf(f3dot(a, a) + EPS12); }
__device__ __forceinline__ float angle3(F3 a, F3 b) {
  F3 c = f3cross(a, b);
  return atan2f(safenorm3(c), f3dot(a, b));
}

// ---------------- 1. Farthest point sampling: 1 block per batch ----------------
#define FPS_T 512
#define FPS_PPT (NN / FPS_T)   // 32 points per thread, kept in VGPRs
__global__ void fps_kernel(const float* __restrict__ xyz, int* __restrict__ fps) {
  const int b = blockIdx.x, tid = threadIdx.x;
  const float* P = xyz + (size_t)b * NN * 3;
  float dmin[FPS_PPT];
#pragma unroll
  for (int q = 0; q < FPS_PPT; ++q) dmin[q] = 1.0e10f;

  __shared__ float swmax[FPS_T / 32];
  __shared__ int   swarg[FPS_T / 32];
  __shared__ int   sfar;
  if (tid == 0) sfar = 0;
  __syncthreads();

  for (int s = 0; s < SS; ++s) {
    int far = sfar;
    if (tid == 0) fps[b * SS + s] = far;   // scan outputs the *old* index
    float cx = P[far * 3 + 0], cy = P[far * 3 + 1], cz = P[far * 3 + 2];
    float bmax = -1.0f; int barg = 0;
#pragma unroll
    for (int q = 0; q < FPS_PPT; ++q) {
      int p3 = (q * FPS_T + tid) * 3;
      float dx = P[p3] - cx, dy = P[p3 + 1] - cy, dz = P[p3 + 2] - cz;
      float dd = dx * dx + dy * dy + dz * dz;
      dmin[q] = fminf(dmin[q], dd);
      if (dmin[q] > bmax) { bmax = dmin[q]; barg = q * FPS_T + tid; }
    }
    for (int off = 16; off > 0; off >>= 1) {    // wave32 argmax reduce
      float om = __shfl_xor(bmax, off, 32);
      int   oa = __shfl_xor(barg, off, 32);
      if (om > bmax || (om == bmax && oa < barg)) { bmax = om; barg = oa; }
    }
    if ((tid & 31) == 0) { swmax[tid >> 5] = bmax; swarg[tid >> 5] = barg; }
    __syncthreads();
    if (tid < 32) {
      float m2 = (tid < (FPS_T / 32)) ? swmax[tid] : -1.0f;
      int   a2 = (tid < (FPS_T / 32)) ? swarg[tid] : 0;
      for (int off = 16; off > 0; off >>= 1) {
        float om = __shfl_xor(m2, off, 32);
        int   oa = __shfl_xor(a2, off, 32);
        if (om > m2 || (om == m2 && oa < a2)) { m2 = om; a2 = oa; }
      }
      if (tid == 0) sfar = a2;
    }
    __syncthreads();
  }
}

// ---------------- 2. gather sampled centers ----------------
__global__ void gather_centers(const float* __restrict__ xyz, const float* __restrict__ normals,
                               const int* __restrict__ fps,
                               float* __restrict__ nxyz, float* __restrict__ nr) {
  int i = blockIdx.x * blockDim.x + threadIdx.x;
  if (i >= BB * SS) return;
  int b = i / SS, j = fps[i];
  const size_t src = ((size_t)b * NN + j) * 3;
#pragma unroll
  for (int d = 0; d < 3; ++d) {
    nxyz[i * 3 + d] = xyz[src + d];
    nr[i * 3 + d]   = normals[src + d];
  }
}

// ---------------- 3. ball query: one wave32 per center, ordered compaction ----------------
__global__ void ball_query(const float* __restrict__ xyz, const float* __restrict__ nxyz,
                           int* __restrict__ idx) {
  int wid  = (blockIdx.x * blockDim.x + threadIdx.x) >> 5;
  int lane = threadIdx.x & 31;
  if (wid >= BB * SS) return;
  int b = wid / SS;
  const float* P = xyz + (size_t)b * NN * 3;
  float cx = nxyz[wid * 3], cy = nxyz[wid * 3 + 1], cz = nxyz[wid * 3 + 2];
  int* o = idx + (size_t)wid * KK;
  int cnt = 0;
  for (int base = 0; base < NN && cnt < KK; base += 32) {
    int j = base + lane;
    float dx = P[j * 3] - cx, dy = P[j * 3 + 1] - cy, dz = P[j * 3 + 2] - cz;
    bool pred = (dx * dx + dy * dy + dz * dz) <= R2;
    unsigned mask = __builtin_amdgcn_ballot_w32(pred);          // wave32 native
    int prefix = __popc(mask & ((1u << lane) - 1u));
    if (pred && (cnt + prefix) < KK) o[cnt + prefix] = j;
    cnt += __popc(mask);
  }
  cnt = cnt > KK ? KK : cnt;
  int first = o[0];              // center itself always in-radius -> slot 0 valid
  for (int t = cnt + lane; t < KK; t += 32) o[t] = first;
}

// ---------------- 4. local PPF features (B,S,K,4) -> bf16, K-padded to 32 ----------------
__global__ void local_feats(const float* __restrict__ xyz, const float* __restrict__ normals,
                            const float* __restrict__ nxyz, const float* __restrict__ nr,
                            const int* __restrict__ idx, __bf16* __restrict__ out) {
  int i = blockIdx.x * blockDim.x + threadIdx.x;
  if (i >= BB * SS * KK) return;
  int bs = i / KK, b = bs / SS;
  F3 c  = ld3(nxyz + (size_t)bs * 3);
  F3 nc = ld3(nr + (size_t)bs * 3);
  int j = idx[i];
  F3 pt = ld3(xyz + ((size_t)b * NN + j) * 3);
  F3 gn = ld3(normals + ((size_t)b * NN + j) * 3);
  F3 d = f3sub(pt, c);
  v8bf z, v0;
#pragma unroll
  for (int e = 0; e < 8; ++e) { z[e] = (__bf16)0.0f; v0[e] = (__bf16)0.0f; }
  v0[0] = (__bf16)angle3(nc, d);
  v0[1] = (__bf16)angle3(gn, d);
  v0[2] = (__bf16)angle3(nc, gn);
  v0[3] = (__bf16)safenorm3(d);
  v8bf* o = (v8bf*)(out + (size_t)i * 32);
  o[0] = v0; o[1] = z; o[2] = z; o[3] = z;
}

// ---------------- 5. global pairwise features (B,S,S,7) -> bf16, K-padded to 32 ----------------
__global__ void global_feats(const float* __restrict__ nxyz, const float* __restrict__ nr,
                             __bf16* __restrict__ out) {
  int i = blockIdx.x * blockDim.x + threadIdx.x;
  if (i >= BB * SS * SS) return;
  int j = i % SS, t = i / SS, ii = t % SS, b = t / SS;
  F3 ci = ld3(nxyz + ((size_t)b * SS + ii) * 3);
  F3 cj = ld3(nxyz + ((size_t)b * SS + j) * 3);
  F3 ni = ld3(nr + ((size_t)b * SS + ii) * 3);
  F3 nj = ld3(nr + ((size_t)b * SS + j) * 3);
  F3 dg = f3sub(ci, cj);
  v8bf z, v0;
#pragma unroll
  for (int e = 0; e < 8; ++e) { z[e] = (__bf16)0.0f; v0[e] = (__bf16)0.0f; }
  v0[0] = (__bf16)angle3(ni, dg);
  v0[1] = (__bf16)angle3(nj, dg);
  v0[2] = (__bf16)angle3(ni, nj);
  v0[3] = (__bf16)safenorm3(dg);
  v0[4] = (__bf16)ci.x; v0[5] = (__bf16)ci.y; v0[6] = (__bf16)ci.z;
  v8bf* o = (v8bf*)(out + (size_t)i * 32);
  o[0] = v0; o[1] = z; o[2] = z; o[3] = z;
}

// ---------------- 6. weight conversion: f32 (Cout,Cin) -> bf16 (Cout,Kpad) ----------------
__global__ void convert_weights(const float* __restrict__ W, __bf16* __restrict__ Wb,
                                int Cout, int Cin, int Kpad) {
  int i = blockIdx.x * blockDim.x + threadIdx.x;
  if (i >= Cout * Kpad) return;
  int o = i / Kpad, k = i % Kpad;
  Wb[i] = (k < Cin) ? (__bf16)W[o * Cin + k] : (__bf16)0.0f;
}

// ---------------- 7. WMMA conv1x1: one wave computes a full 16 x COUT strip ----------------
// A fragment (CDNA5 bf16 layout): lane elements 0-7 = K half*8+0..7, 8-15 = K 16+half*8+0..7
//  -> two contiguous b128 loads.  B fragment: same K layout as A -> two b128 loads.
// K fully unrolled (CIN multiple of 32); A fragment reused across all NT n-tiles.
template<int CIN, int NT>
__global__ void gemm_wmma(const __bf16* __restrict__ A,    // rows x CIN (padded, zero-filled)
                          const __bf16* __restrict__ Wb,   // (NT*16) x CIN bf16
                          const float* __restrict__ bias,
                          __bf16* __restrict__ Y,          // rows x (NT*16)
                          float* __restrict__ chanSum, float* __restrict__ chanSqs,
                          int rows, int rowsPerBatch) {
  constexpr int COUT = NT * 16;
  const int lane = threadIdx.x & 31;
  const int mt = blockIdx.x * (blockDim.x >> 5) + (threadIdx.x >> 5);
  if (mt >= (rows >> 4)) return;
  const int m0 = mt << 4;
  const int half = lane >> 4;
  const int r16  = lane & 15;

  v8f zero = {};
  v8f acc[NT];
#pragma unroll
  for (int nt = 0; nt < NT; ++nt) acc[nt] = zero;

  const __bf16* Arow = A + (size_t)(m0 + r16) * CIN;
#pragma unroll
  for (int ks = 0; ks < CIN / 32; ++ks) {
    const int k0 = ks * 32;
    v8bf a0 = *(const v8bf*)(Arow + k0 + half * 8);
    v8bf a1 = *(const v8bf*)(Arow + k0 + 16 + half * 8);
    v16bf a;
#pragma unroll
    for (int e = 0; e < 8; ++e) { a[e] = a0[e]; a[8 + e] = a1[e]; }
#pragma unroll
    for (int nt = 0; nt < NT; ++nt) {
      const __bf16* Wp = Wb + (size_t)(nt * 16 + r16) * CIN + k0 + half * 8;
      v8bf b0 = *(const v8bf*)(Wp);
      v8bf b1 = *(const v8bf*)(Wp + 16);
      v16bf bm;
#pragma unroll
      for (int e = 0; e < 8; ++e) { bm[e] = b0[e]; bm[8 + e] = b1[e]; }
      acc[nt] = __builtin_amdgcn_wmma_f32_16x16x32_bf16(false, a, false, bm,
                                                        (short)0, acc[nt], false, false);
    }
  }

  const int b = m0 / rowsPerBatch;
#pragma unroll
  for (int nt = 0; nt < NT; ++nt) {
    int n = nt * 16 + r16;
    float bv = bias[n];
    float lsum = 0.0f, lsq = 0.0f;
#pragma unroll
    for (int r = 0; r < 8; ++r) {       // C/D: VGPR r -> M = r + half*8, N = lane&15
      int m = m0 + r + half * 8;
      float y = acc[nt][r] + bv;
      Y[(size_t)m * COUT + n] = (__bf16)y;
      lsum += y; lsq += y * y;
    }
    if (chanSum) {
      lsum += __shfl_xor(lsum, 16, 32);
      lsq  += __shfl_xor(lsq, 16, 32);
      if (half == 0) {
        atomicAdd(&chanSum[b * COUT + n], lsum);
        atomicAdd(&chanSqs[b * COUT + n], lsq);
      }
    }
  }
}

// ---------------- 8. GroupNorm stats + normalize ----------------
__global__ void gn_stats(const float* __restrict__ chanSum, const float* __restrict__ chanSqs,
                         float* __restrict__ stats, int Cout, int rowsPerBatch) {
  int i = blockIdx.x * blockDim.x + threadIdx.x;     // b*8 + g
  if (i >= BB * 8) return;
  int b = i >> 3, g = i & 7, Cg = Cout >> 3;
  float s = 0.0f, q = 0.0f;
  for (int cc = g * Cg; cc < (g + 1) * Cg; ++cc) {
    s += chanSum[b * Cout + cc];
    q += chanSqs[b * Cout + cc];
  }
  float cnt = (float)rowsPerBatch * (float)Cg;
  float m = s / cnt;
  float var = q / cnt - m * m;
  stats[i * 2]     = m;
  stats[i * 2 + 1] = rsqrtf(var + 1e-5f);
}

__global__ void gn_norm_relu(__bf16* __restrict__ Y, const float* __restrict__ stats,
                             const float* __restrict__ gamma, const float* __restrict__ beta,
                             int Cout, size_t total, int rowsPerBatch) {
  size_t i = (size_t)blockIdx.x * blockDim.x + threadIdx.x;
  if (i >= total) return;
  int cc = (int)(i % Cout);
  int b  = (int)((i / Cout) / rowsPerBatch);
  int g  = cc / (Cout >> 3);
  float m = stats[(b * 8 + g) * 2], inv = stats[(b * 8 + g) * 2 + 1];
  float y = ((float)Y[i] - m) * inv * gamma[cc] + beta[cc];
  Y[i] = (__bf16)fmaxf(y, 0.0f);
}

// ---------------- 9. max pooling ----------------
__global__ void maxpool_local(const __bf16* __restrict__ X, __bf16* __restrict__ Yo) {
  int i = blockIdx.x * blockDim.x + threadIdx.x;
  if (i >= BB * SS * C2) return;
  int cc = i % C2, bs = i / C2;
  const __bf16* base = X + (size_t)bs * KK * C2 + cc;
  float m = -3.0e38f;
  for (int k = 0; k < KK; ++k) m = fmaxf(m, (float)base[(size_t)k * C2]);
  Yo[i] = (__bf16)m;
}

__global__ void maxpool_global(const __bf16* __restrict__ X, __bf16* __restrict__ Yo) {
  int i = blockIdx.x * blockDim.x + threadIdx.x;
  if (i >= BB * SS * C2) return;
  int cc = i % C2, bsj = i / C2, b = bsj / SS, j = bsj % SS;
  const __bf16* base = X + (((size_t)b * SS) * SS + j) * C2 + cc;  // max over axis i
  float m = -3.0e38f;
  for (int ii = 0; ii < SS; ++ii) m = fmaxf(m, (float)base[(size_t)ii * SS * C2]);
  Yo[i] = (__bf16)m;
}

// ---------------- 10. finalize: L2-normalize xl,xg; emit outputs ----------------
__global__ void finalize_out(const __bf16* __restrict__ xl, const __bf16* __restrict__ xg,
                             const float* __restrict__ nxyz, const float* __restrict__ nr,
                             float* __restrict__ out) {
  int row = blockIdx.x * blockDim.x + threadIdx.x;
  if (row >= BB * SS) return;
  float sl = 0.0f, sg = 0.0f;
  for (int cc = 0; cc < EMB; ++cc) {
    float a = (float)xl[(size_t)row * EMB + cc]; sl += a * a;
    float g = (float)xg[(size_t)row * EMB + cc]; sg += g * g;
  }
  float il = 1.0f / sqrtf(sl + EPS12);
  float ig = 1.0f / sqrtf(sg + EPS12);
  for (int cc = 0; cc < EMB; ++cc)
    out[(size_t)row * EMB + cc] =
        (float)xl[(size_t)row * EMB + cc] * il + (float)xg[(size_t)row * EMB + cc] * ig;
  const int base1 = BB * SS * EMB;
#pragma unroll
  for (int d = 0; d < 3; ++d) {
    out[base1 + row * 3 + d] = nxyz[row * 3 + d];
    out[base1 + BB * SS * 3 + row * 3 + d] = nr[row * 3 + d];
  }
}

// ---------------- host ----------------
static inline char* ws_bump(char*& p, size_t bytes) {
  char* r = p;
  p += (bytes + 255) & ~(size_t)255;
  return r;
}

extern "C" void kernel_launch(void* const* d_in, const int* in_sizes, int n_in,
                              void* d_out, int out_size, void* d_ws, size_t ws_size,
                              hipStream_t stream) {
  (void)in_sizes; (void)n_in; (void)out_size; (void)ws_size;
  const float* xyz     = (const float*)d_in[0];
  const float* normals = (const float*)d_in[1];
  auto P = [&](int i) { return (const float*)d_in[i]; };
  // param index map (jax pytree flattening of the setup dict):
  // pre_l: 2..13 (3x{W,b,gamma,beta}), post_l: 14..23 (4,4,{W,b}),
  // pre_g: 24..35, post_g: 36..45

  char* pws = (char*)d_ws;
  int*    fps    = (int*)   ws_bump(pws, (size_t)BB * SS * 4);
  float*  nxyz   = (float*) ws_bump(pws, (size_t)BB * SS * 3 * 4);
  float*  nr     = (float*) ws_bump(pws, (size_t)BB * SS * 3 * 4);
  int*    bidx   = (int*)   ws_bump(pws, (size_t)BB * SS * KK * 4);
  __bf16* localF = (__bf16*)ws_bump(pws, (size_t)BB * SS * KK * 32 * 2);   // K-padded
  __bf16* gfeat  = (__bf16*)ws_bump(pws, (size_t)BB * SS * SS * 32 * 2);   // K-padded
  const size_t bigRows = (size_t)BB * SS * SS;           // 524288
  __bf16* bufA   = (__bf16*)ws_bump(pws, bigRows * C2 * 2);
  __bf16* bufB   = (__bf16*)ws_bump(pws, bigRows * C2 * 2);
  __bf16* poolL  = (__bf16*)ws_bump(pws, (size_t)BB * SS * C2 * 2);
  __bf16* poolG  = (__bf16*)ws_bump(pws, (size_t)BB * SS * C2 * 2);
  __bf16* postA  = (__bf16*)ws_bump(pws, (size_t)BB * SS * C2 * 2);
  __bf16* postB  = (__bf16*)ws_bump(pws, (size_t)BB * SS * C2 * 2);
  __bf16* xl     = (__bf16*)ws_bump(pws, (size_t)BB * SS * EMB * 2);
  __bf16* xg     = (__bf16*)ws_bump(pws, (size_t)BB * SS * EMB * 2);
  float*  chanSum = (float*)ws_bump(pws, (size_t)2 * BB * C2 * 4);
  float*  chanSqs = chanSum + BB * C2;
  float*  stats   = (float*)ws_bump(pws, (size_t)BB * 8 * 2 * 4);
  __bf16* wbArena = (__bf16*)ws_bump(pws, (size_t)12 * C2 * C2 * 2);  // 12 layers max size
  int     wbNext  = 0;

  // conv1x1 + optional GroupNorm/ReLU; Kpad = CIN template (Cin rounded up to 32)
  auto conv_gn = [&](const __bf16* Ain, int Cin, int Kpad, int wi, __bf16* Yout,
                     size_t rows, int Cout, int rpb, bool gn) {
    __bf16* Wb = wbArena + (size_t)(wbNext++) * C2 * C2;
    int wtot = Cout * Kpad;
    convert_weights<<<(wtot + 255) / 256, 256, 0, stream>>>(P(wi), Wb, Cout, Cin, Kpad);
    if (gn) hipMemsetAsync(chanSum, 0, 2 * BB * C2 * sizeof(float), stream);
    float* cs = gn ? chanSum : nullptr;
    float* cq = gn ? chanSqs : nullptr;
    int blocks = (int)((rows / 16 + 3) / 4);
    if (Kpad == 32 && Cout == 96)
      gemm_wmma<32, 6><<<blocks, 128, 0, stream>>>(Ain, Wb, P(wi + 1), Yout, cs, cq, (int)rows, rpb);
    else if (Kpad == 96 && Cout == 96)
      gemm_wmma<96, 6><<<blocks, 128, 0, stream>>>(Ain, Wb, P(wi + 1), Yout, cs, cq, (int)rows, rpb);
    else if (Kpad == 96 && Cout == 192)
      gemm_wmma<96, 12><<<blocks, 128, 0, stream>>>(Ain, Wb, P(wi + 1), Yout, cs, cq, (int)rows, rpb);
    else if (Kpad == 192 && Cout == 192)
      gemm_wmma<192, 12><<<blocks, 128, 0, stream>>>(Ain, Wb, P(wi + 1), Yout, cs, cq, (int)rows, rpb);
    else // Kpad == 192 && Cout == 96
      gemm_wmma<192, 6><<<blocks, 128, 0, stream>>>(Ain, Wb, P(wi + 1), Yout, cs, cq, (int)rows, rpb);
    if (gn) {
      gn_stats<<<1, 64, 0, stream>>>(chanSum, chanSqs, stats, Cout, rpb);
      size_t total = rows * (size_t)Cout;
      gn_norm_relu<<<(int)((total + 255) / 256), 256, 0, stream>>>(
          Yout, stats, P(wi + 2), P(wi + 3), Cout, total, rpb);
    }
  };

  // geometry
  fps_kernel<<<BB, FPS_T, 0, stream>>>(xyz, fps);
  gather_centers<<<(BB * SS + 255) / 256, 256, 0, stream>>>(xyz, normals, fps, nxyz, nr);
  ball_query<<<(BB * SS * 32 + 255) / 256, 256, 0, stream>>>(xyz, nxyz, bidx);
  local_feats<<<(BB * SS * KK + 255) / 256, 256, 0, stream>>>(xyz, normals, nxyz, nr, bidx, localF);
  global_feats<<<(BB * SS * SS + 255) / 256, 256, 0, stream>>>(nxyz, nr, gfeat);

  // local prepool 4 -> 96 -> 96 -> 192, GN over S*K per batch
  conv_gn(localF, 4, 32,  2,  bufA, (size_t)BB * SS * KK, EMB, SS * KK, true);
  conv_gn(bufA, 96, 96,   6,  bufB, (size_t)BB * SS * KK, EMB, SS * KK, true);
  conv_gn(bufB, 96, 96,   10, bufA, (size_t)BB * SS * KK, C2,  SS * KK, true);
  maxpool_local<<<(BB * SS * C2 + 255) / 256, 256, 0, stream>>>(bufA, poolL);

  // global prepool 7 -> 96 -> 96 -> 192, GN over S*S per batch
  conv_gn(gfeat, 7, 32,  24, bufA, bigRows, EMB, SS * SS, true);
  conv_gn(bufA, 96, 96,  28, bufB, bigRows, EMB, SS * SS, true);
  conv_gn(bufB, 96, 96,  32, bufA, bigRows, C2,  SS * SS, true);
  maxpool_global<<<(BB * SS * C2 + 255) / 256, 256, 0, stream>>>(bufA, poolG);

  // postpool local: 192 -> 192 -> 96 -> 96(final, no GN)
  conv_gn(poolL, 192, 192, 14, postA, (size_t)BB * SS, C2,  SS, true);
  conv_gn(postA, 192, 192, 18, postB, (size_t)BB * SS, EMB, SS, true);
  conv_gn(postB, 96, 96,   22, xl,    (size_t)BB * SS, EMB, SS, false);

  // postpool global
  conv_gn(poolG, 192, 192, 36, postA, (size_t)BB * SS, C2,  SS, true);
  conv_gn(postA, 192, 192, 40, postB, (size_t)BB * SS, EMB, SS, true);
  conv_gn(postB, 96, 96,   44, xg,    (size_t)BB * SS, EMB, SS, false);

  finalize_out<<<(BB * SS + 255) / 256, 256, 0, stream>>>(xl, xg, nxyz, nr, (float*)d_out);
}
